// MultiHeadAttentionWithLoRA_36971078484329
// MI455X (gfx1250) — matrix operand order repaired and checked by reading, not compiled, not run on hardware
//
#include <hip/hip_runtime.h>

typedef __attribute__((ext_vector_type(16))) _Float16 v16h;
typedef __attribute__((ext_vector_type(8)))  float    v8f;
typedef int v4i_vs __attribute__((vector_size(16)));   // matches builtin's V4i

#define D   768
#define RR  8
#define H   12
#define DH  64
#define SS  2048
#define MM  8192   // B*S

union Frag {
  v16h  h;
  uint4 q[2];
};

__device__ __forceinline__ v8f wmma_f16(const Frag& a, const Frag& b, v8f c) {
  // D = A(16x32 f16) * B(32x16 f16) + C(16x16 f32)
  return __builtin_amdgcn_wmma_f32_16x16x32_f16(false, a.h, false, b.h,
                                                (short)0, c, false, false);
}

// -------- CDNA5 async global->LDS copy (16B), with safe fallback ------------
#if __has_builtin(__builtin_amdgcn_global_load_async_to_lds_b128)
__device__ __forceinline__ void async_cp16(_Float16* lds_dst, const _Float16* g_src) {
  __builtin_amdgcn_global_load_async_to_lds_b128(
      (__attribute__((address_space(1))) v4i_vs*)(g_src),
      (__attribute__((address_space(3))) v4i_vs*)(lds_dst), 0, 0);
}
#define ASYNC_WAIT(imm) asm volatile("s_wait_asynccnt " #imm ::: "memory")
#else
__device__ __forceinline__ void async_cp16(_Float16* lds_dst, const _Float16* g_src) {
  *(uint4*)lds_dst = *(const uint4*)g_src;
}
#define ASYNC_WAIT(imm) asm volatile("" ::: "memory")
#endif

// ---------------------------------------------------------------- prep kernels

__global__ void k_f32_to_f16(const float* __restrict__ src,
                             _Float16* __restrict__ dst, int n) {
  int i = blockIdx.x * 256 + threadIdx.x;
  if (i < n) dst[i] = (_Float16)src[i];
}

// weffT[mat][n][k] = W0[k][n] + sum_r A[k][r]*Bm[r][n]   (transposed, f16)
__global__ void k_build_weffT(const float* __restrict__ Wq0, const float* __restrict__ Aq, const float* __restrict__ Bq,
                              const float* __restrict__ Wk0, const float* __restrict__ Ak, const float* __restrict__ Bk,
                              const float* __restrict__ Wv0, const float* __restrict__ Av, const float* __restrict__ Bv,
                              _Float16* __restrict__ weffT) {
  int i = blockIdx.x * 256 + threadIdx.x;     // ((mat*768)+n)*768 + k
  int k = i % D;
  int t = i / D;
  int n = t % D;
  int mat = t / D;
  const float *W0, *A, *Bm;
  if (mat == 0)      { W0 = Wq0; A = Aq; Bm = Bq; }
  else if (mat == 1) { W0 = Wk0; A = Ak; Bm = Bk; }
  else               { W0 = Wv0; A = Av; Bm = Bv; }
  float acc = W0[k * D + n];
#pragma unroll
  for (int r = 0; r < RR; ++r) acc += A[k * RR + r] * Bm[r * D + n];
  weffT[i] = (_Float16)acc;
}

// woT[n][k] = Wo[k][n]   (f16)
__global__ void k_build_woT(const float* __restrict__ Wo, _Float16* __restrict__ woT) {
  int i = blockIdx.x * 256 + threadIdx.x;
  int k = i % D, n = i / D;
  woT[i] = (_Float16)Wo[k * D + n];
}

// -------------------------------------------------- QKV projection (3 GEMMs)
// Block = 128x64 tile; weight k-chunk staged in LDS (async, double-buffered).
__global__ void __launch_bounds__(256) k_qkv_gemm(const _Float16* __restrict__ xb,
                                                  const _Float16* __restrict__ weffT,
                                                  _Float16* __restrict__ qb,
                                                  _Float16* __restrict__ kb,
                                                  _Float16* __restrict__ vT) {
  __shared__ _Float16 bTile[2][64 * 32];      // [col 0..63][k 0..31]
  const int tid = threadIdx.x;
  const int wave = tid >> 5, lane = tid & 31;
  const int l = lane & 15, hi = lane >> 4;
  int mat = blockIdx.x / 768;                 // 3 * 64 * 12 = 2304 blocks
  int rem = blockIdx.x % 768;
  int mt  = rem / 12;                         // 128-row tile (0..63)
  int nt  = rem % 12;                         // 64-col tile
  const _Float16* wbase = weffT + (size_t)mat * D * D + (size_t)(nt * 64) * D;
  const _Float16* arow  = xb + (size_t)(mt * 128 + wave * 16 + l) * D;

  // cooperative copy: thread -> one 16B piece of the 64x32 weight chunk
  const int bcol = tid >> 2, bpart = tid & 3;
  const _Float16* gsrc = wbase + (size_t)bcol * D + bpart * 8;
  const int ldst = bcol * 32 + bpart * 8;

  async_cp16(&bTile[0][ldst], gsrc);
  v8f acc[4] = {};
  for (int i = 0; i < 24; ++i) {
    int k0 = i * 32;
    if (i + 1 < 24) {
      async_cp16(&bTile[(i + 1) & 1][ldst], gsrc + k0 + 32);
      ASYNC_WAIT(0x1);
    } else {
      ASYNC_WAIT(0x0);
    }
    __syncthreads();                          // chunk i visible to all waves
    Frag a;
    a.q[0] = *(const uint4*)(arow + k0 + 16 * hi);
    a.q[1] = *(const uint4*)(arow + k0 + 8 + 16 * hi);
    const _Float16* bt = bTile[i & 1];
#pragma unroll
    for (int f = 0; f < 4; ++f) {
      Frag b;
      const _Float16* bp = bt + (f * 16 + l) * 32 + 16 * hi;
      b.q[0] = ((const uint4*)bp)[0];
      b.q[1] = ((const uint4*)bp)[1];
      acc[f] = wmma_f16(a, b, acc[f]);
    }
    __syncthreads();                          // done reading before overwrite
  }
#pragma unroll
  for (int f = 0; f < 4; ++f) {
    int n = nt * 64 + f * 16 + l;
    int h = n >> 6, d = n & 63;
#pragma unroll
    for (int j = 0; j < 8; ++j) {
      int m  = mt * 128 + wave * 16 + j + 8 * hi;
      int b_ = m >> 11;
      int s  = m & 2047;
      _Float16 val = (_Float16)acc[f][j];
      size_t hb = (size_t)(b_ * H + h);
      if (mat == 0)      qb[(hb * SS + s) * DH + d] = val;
      else if (mat == 1) kb[(hb * SS + s) * DH + d] = val;
      else               vT[(hb * DH + d) * SS + s] = val;
    }
  }
}

// --------------------------------------------- flash attention
// Block = 8 q-tiles of one (b,h); K/V chunks shared via LDS (async, dbl-buffered)
__global__ void __launch_bounds__(256) k_attn(const _Float16* __restrict__ qb,
                                              const _Float16* __restrict__ kb,
                                              const _Float16* __restrict__ vT,
                                              _Float16* __restrict__ cc) {
  __shared__ _Float16 kTile[2][32 * 64];      // [key row 0..31][d 0..63]
  __shared__ _Float16 vTile[2][64 * 32];      // [d 0..63][key 0..31]
  __shared__ _Float16 pT[8][16 * 32];         // per-wave P transpose
  const int tid = threadIdx.x;
  const int wave = tid >> 5, lane = tid & 31;
  const int l = lane & 15, hi = lane >> 4;
  int bh = blockIdx.x / 16;                   // same (b,h) for whole block
  int qt = (blockIdx.x % 16) * 8 + wave;
  const _Float16* qrow  = qb + ((size_t)bh * SS + qt * 16 + l) * DH;
  const _Float16* kbase = kb + (size_t)bh * SS * DH;
  const _Float16* vbase = vT + (size_t)bh * DH * SS;
  _Float16* pw = pT[wave];

  // cooperative copy indices: 4KB K chunk + 4KB V chunk, 16B per thread each
  const int krow = tid >> 3, kpart = tid & 7;
  const _Float16* kg = kbase + (size_t)krow * DH + kpart * 8;
  const int kld = krow * 64 + kpart * 8;
  const int vrow = tid >> 2, vpart = tid & 3;
  const _Float16* vg = vbase + (size_t)vrow * SS + vpart * 8;
  const int vld = vrow * 32 + vpart * 8;

  Frag qa0, qa1;                              // Q rows, d=0..31 / 32..63
  qa0.q[0] = *(const uint4*)(qrow + 16 * hi);
  qa0.q[1] = *(const uint4*)(qrow + 8 + 16 * hi);
  qa1.q[0] = *(const uint4*)(qrow + 32 + 16 * hi);
  qa1.q[1] = *(const uint4*)(qrow + 40 + 16 * hi);

  v8f o[4] = {};
  float mrun[8], lrun[8];
#pragma unroll
  for (int j = 0; j < 8; ++j) { mrun[j] = -1e30f; lrun[j] = 0.f; }

  async_cp16(&kTile[0][kld], kg);
  async_cp16(&vTile[0][vld], vg);
  for (int i = 0; i < 64; ++i) {
    int n0 = i * 32;
    if (i + 1 < 64) {
      async_cp16(&kTile[(i + 1) & 1][kld], kg + (size_t)(n0 + 32) * DH);
      async_cp16(&vTile[(i + 1) & 1][vld], vg + n0 + 32);
      ASYNC_WAIT(0x2);
    } else {
      ASYNC_WAIT(0x0);
    }
    __syncthreads();
    const _Float16* kt = kTile[i & 1];
    const _Float16* vt = vTile[i & 1];

    v8f s1 = {}, s2 = {};
    {
      Frag b;
      const _Float16* kr1 = kt + l * 64 + 16 * hi;
      const _Float16* kr2 = kt + (16 + l) * 64 + 16 * hi;
      b.q[0] = ((const uint4*)kr1)[0];        b.q[1] = ((const uint4*)kr1)[1];
      s1 = wmma_f16(qa0, b, s1);
      b.q[0] = ((const uint4*)(kr1 + 32))[0]; b.q[1] = ((const uint4*)(kr1 + 32))[1];
      s1 = wmma_f16(qa1, b, s1);
      b.q[0] = ((const uint4*)kr2)[0];        b.q[1] = ((const uint4*)kr2)[1];
      s2 = wmma_f16(qa0, b, s2);
      b.q[0] = ((const uint4*)(kr2 + 32))[0]; b.q[1] = ((const uint4*)(kr2 + 32))[1];
      s2 = wmma_f16(qa1, b, s2);
    }
    // online softmax: rows j+8*hi live across the 16 lanes of each half-wave
    float mt_[8];
#pragma unroll
    for (int j = 0; j < 8; ++j) {
      s1[j] *= 0.125f; s2[j] *= 0.125f;       // 1/sqrt(64)
      mt_[j] = fmaxf(s1[j], s2[j]);
    }
#pragma unroll
    for (int mask = 1; mask < 16; mask <<= 1)
#pragma unroll
      for (int j = 0; j < 8; ++j) mt_[j] = fmaxf(mt_[j], __shfl_xor(mt_[j], mask, 16));

    float rs[8];
#pragma unroll
    for (int j = 0; j < 8; ++j) {
      float mn = fmaxf(mrun[j], mt_[j]);
      float al = __expf(mrun[j] - mn);
      mrun[j] = mn;
      s1[j] = __expf(s1[j] - mn);
      s2[j] = __expf(s2[j] - mn);
      rs[j] = s1[j] + s2[j];
      lrun[j] *= al;
      o[0][j] *= al; o[1][j] *= al; o[2][j] *= al; o[3][j] *= al;
    }
#pragma unroll
    for (int mask = 1; mask < 16; mask <<= 1)
#pragma unroll
      for (int j = 0; j < 8; ++j) rs[j] += __shfl_xor(rs[j], mask, 16);
#pragma unroll
    for (int j = 0; j < 8; ++j) lrun[j] += rs[j];

    // C-layout -> A-layout transpose of P through per-wave LDS
#pragma unroll
    for (int j = 0; j < 8; ++j) {
      int row = j + 8 * hi;
      pw[row * 32 + l]      = (_Float16)s1[j];
      pw[row * 32 + 16 + l] = (_Float16)s2[j];
    }
    asm volatile("s_wait_dscnt 0" ::: "memory");   // LDS RAW fence
    Frag pa;
    pa.q[0] = *(const uint4*)(pw + l * 32 + 16 * hi);
    pa.q[1] = *(const uint4*)(pw + l * 32 + 8 + 16 * hi);
#pragma unroll
    for (int f = 0; f < 4; ++f) {
      Frag vb;
      const _Float16* vr = vt + (f * 16 + l) * 32 + 16 * hi;
      vb.q[0] = ((const uint4*)vr)[0];
      vb.q[1] = ((const uint4*)vr)[1];
      o[f] = wmma_f16(pa, vb, o[f]);
    }
    __syncthreads();                          // done reading before overwrite
  }

  int h  = bh % H;
  int b_ = bh / H;
  float inv[8];
#pragma unroll
  for (int j = 0; j < 8; ++j) inv[j] = 1.0f / lrun[j];
#pragma unroll
  for (int f = 0; f < 4; ++f)
#pragma unroll
    for (int j = 0; j < 8; ++j) {
      int row = b_ * SS + qt * 16 + j + 8 * hi;
      int col = h * DH + f * 16 + l;
      cc[(size_t)row * D + col] = (_Float16)(o[f][j] * inv[j]);
    }
}

// ------------------------------------------------ output projection + bias
__global__ void __launch_bounds__(256) k_out_gemm(const _Float16* __restrict__ cc,
                                                  const _Float16* __restrict__ woT,
                                                  const float* __restrict__ bo,
                                                  float* __restrict__ out) {
  __shared__ _Float16 bTile[2][64 * 32];
  const int tid = threadIdx.x;
  const int wave = tid >> 5, lane = tid & 31;
  const int l = lane & 15, hi = lane >> 4;
  int mt = blockIdx.x / 12;                   // 64 * 12 = 768 blocks
  int nt = blockIdx.x % 12;
  const _Float16* wbase = woT + (size_t)(nt * 64) * D;
  const _Float16* arow  = cc + (size_t)(mt * 128 + wave * 16 + l) * D;

  const int bcol = tid >> 2, bpart = tid & 3;
  const _Float16* gsrc = wbase + (size_t)bcol * D + bpart * 8;
  const int ldst = bcol * 32 + bpart * 8;

  async_cp16(&bTile[0][ldst], gsrc);
  v8f acc[4] = {};
  for (int i = 0; i < 24; ++i) {
    int k0 = i * 32;
    if (i + 1 < 24) {
      async_cp16(&bTile[(i + 1) & 1][ldst], gsrc + k0 + 32);
      ASYNC_WAIT(0x1);
    } else {
      ASYNC_WAIT(0x0);
    }
    __syncthreads();
    Frag a;
    a.q[0] = *(const uint4*)(arow + k0 + 16 * hi);
    a.q[1] = *(const uint4*)(arow + k0 + 8 + 16 * hi);
    const _Float16* bt = bTile[i & 1];
#pragma unroll
    for (int f = 0; f < 4; ++f) {
      Frag b;
      const _Float16* bp = bt + (f * 16 + l) * 32 + 16 * hi;
      b.q[0] = ((const uint4*)bp)[0];
      b.q[1] = ((const uint4*)bp)[1];
      acc[f] = wmma_f16(a, b, acc[f]);
    }
    __syncthreads();
  }
#pragma unroll
  for (int f = 0; f < 4; ++f) {
    int n = nt * 64 + f * 16 + l;
    float bias = bo[n];
#pragma unroll
    for (int j = 0; j < 8; ++j) {
      int m = mt * 128 + wave * 16 + j + 8 * hi;
      out[(size_t)m * D + n] = acc[f][j] + bias;
    }
  }
}

// ---------------------------------------------------------------------------

extern "C" void kernel_launch(void* const* d_in, const int* in_sizes, int n_in,
                              void* d_out, int out_size, void* d_ws, size_t ws_size,
                              hipStream_t stream) {
  const float* x   = (const float*)d_in[0];
  const float* Wq0 = (const float*)d_in[1];
  const float* Aq  = (const float*)d_in[2];
  const float* Bq  = (const float*)d_in[3];
  const float* Wk0 = (const float*)d_in[4];
  const float* Ak  = (const float*)d_in[5];
  const float* Bk  = (const float*)d_in[6];
  const float* Wv0 = (const float*)d_in[7];
  const float* Av  = (const float*)d_in[8];
  const float* Bv  = (const float*)d_in[9];
  const float* Wo  = (const float*)d_in[10];
  const float* bo  = (const float*)d_in[11];
  float* out = (float*)d_out;

  // workspace layout (f16), ~68 MB total
  _Float16* ws    = (_Float16*)d_ws;
  const size_t NXB = (size_t)MM * D;           // 6,291,456 elements
  _Float16* xb    = ws;
  _Float16* weffT = xb + NXB;                  // [3][768][768]
  _Float16* woT   = weffT + (size_t)3 * D * D;
  _Float16* qb    = woT + (size_t)D * D;       // [B*H][S][Dh]
  _Float16* kb    = qb + NXB;                  // [B*H][S][Dh]
  _Float16* vT    = kb + NXB;                  // [B*H][Dh][S]
  _Float16* cc    = vT + NXB;                  // [B*S][D]

  k_f32_to_f16 <<<(int)(NXB / 256), 256, 0, stream>>>(x, xb, (int)NXB);
  k_build_weffT<<<3 * D * D / 256, 256, 0, stream>>>(Wq0, Aq, Bq, Wk0, Ak, Bk,
                                                     Wv0, Av, Bv, weffT);
  k_build_woT  <<<D * D / 256, 256, 0, stream>>>(Wo, woT);
  k_qkv_gemm   <<<2304, 256, 0, stream>>>(xb, weffT, qb, kb, vT);
  k_attn       <<<768, 256, 0, stream>>>(qb, kb, vT, cc);
  k_out_gemm   <<<768, 256, 0, stream>>>(cc, woT, bo, out);
}
